// Model_57071525429488
// MI455X (gfx1250) — matrix operand order repaired and checked by reading, not compiled, not run on hardware
//
#include <hip/hip_runtime.h>

// ---------------------------------------------------------------------------
// Problem constants (match reference)
// ---------------------------------------------------------------------------
#define N_U 100000
#define N_M 20000
#define N_E 2000000
#define N_L 500000
#define F_U 64
#define F_M 128
#define HH  128
#define DEC_M 64   // edges per decoder block (4 WMMA M-tiles)

typedef __attribute__((ext_vector_type(16))) __bf16 v16bf;
typedef __attribute__((ext_vector_type(8)))  float  v8f;

#define WMMA_BF16(a, b, c) \
  __builtin_amdgcn_wmma_f32_16x16x32_bf16(false, (a), false, (b), (short)0, (c), false, false)

// ---------------------------------------------------------------------------
// WMMA fragment loaders (this code's contiguous k order, not AMD's table order; wave32)
// A 16x32 bf16: lane m = lane&15; lane<16 -> K 0..15, lane>=16 -> K 16..31
// B 32x16 bf16 (B[k][n] = W[n][k]): lane n = lane&15; lane<16 -> K 0..15,
//   lane>=16 -> K 16..31 (contiguous along a W row)
// ---------------------------------------------------------------------------
__device__ __forceinline__ v16bf load_a_frag_lds(const __bf16* __restrict__ base,
                                                 int stride, int lane, int kbase) {
  const int m  = lane & 15;
  const int hb = (lane & 16) ? 16 : 0;
  const __bf16* p = base + m * stride + kbase + hb;
  v16bf a;
#pragma unroll
  for (int t = 0; t < 16; ++t) a[t] = p[t];
  return a;
}

__device__ __forceinline__ v16bf load_b_frag_row(const __bf16* __restrict__ wrow,
                                                 int lane, int kbase) {
  const int hb = (lane & 16) ? 16 : 0;
  const __bf16* q = wrow + kbase + hb;
  v16bf b;
#pragma unroll
  for (int t = 0; t < 16; ++t) b[t] = q[t];
  return b;
}

// ---------------------------------------------------------------------------
// Utility kernels
// ---------------------------------------------------------------------------
__global__ void zero_f32(float* __restrict__ p, long long n) {
  long long i = (long long)blockIdx.x * blockDim.x + threadIdx.x;
  if (i < n) p[i] = 0.0f;
}

__global__ void cvt_f32_bf16(const float* __restrict__ s, __bf16* __restrict__ d, int n) {
  int i = blockIdx.x * blockDim.x + threadIdx.x;
  if (i < n) d[i] = (__bf16)s[i];
}

// ---------------------------------------------------------------------------
// Per-type projection + BatchNorm(eval) + ReLU.  One node per block,
// 128 threads = one output channel each, x-row staged in LDS.
// ---------------------------------------------------------------------------
__global__ void proj_bn_relu(const float* __restrict__ X, const float* __restrict__ W,
                             const float* __restrict__ bias,
                             const float* __restrict__ g, const float* __restrict__ bb,
                             const float* __restrict__ mean, const float* __restrict__ var,
                             float* __restrict__ out, int F) {
  __shared__ float sx[HH];
  const int node = blockIdx.x;
  const int o = threadIdx.x;  // 0..127
  if (o < F) sx[o] = X[(long long)node * F + o];
  __syncthreads();
  float acc = bias[o];
  const float* wr = W + o * F;
  for (int k = 0; k < F; ++k) acc = fmaf(sx[k], wr[k], acc);
  float y = (acc - mean[o]) * rsqrtf(var[o] + 1e-5f) * g[o] + bb[o];
  out[(long long)node * HH + o] = fmaxf(y, 0.0f);
}

// ---------------------------------------------------------------------------
// Edge scatter for segment-mean: 32 threads/edge, float4 per thread.
// ---------------------------------------------------------------------------
__global__ void scatter_edges(const float* __restrict__ xsrc,
                              const int* __restrict__ src, const int* __restrict__ dst,
                              float* __restrict__ agg, float* __restrict__ cnt, int nE) {
  long long idx = (long long)blockIdx.x * blockDim.x + threadIdx.x;
  if (idx >= (long long)nE * 32) return;
  const int e = (int)(idx >> 5);
  const int c = ((int)idx & 31) * 4;
  const int s = src[e], d = dst[e];
  const float4 v = *reinterpret_cast<const float4*>(xsrc + (long long)s * HH + c);
  float* ap = agg + (long long)d * HH + c;
  atomicAdd(ap + 0, v.x); atomicAdd(ap + 1, v.y);
  atomicAdd(ap + 2, v.z); atomicAdd(ap + 3, v.w);
  if (c == 0) atomicAdd(&cnt[d], 1.0f);
}

// ---------------------------------------------------------------------------
// SAGE combine:  out = act( (agg/max(cnt,1)) @ Wl^T + bl + X @ Wr^T )
// 16 rows per block, 8 waves = 8 column tiles of 16.  K = 128 -> 4 k-steps
// per operand.  Conv1 writes f32 (+ReLU); conv2 writes bf16 (decoder input).
// ---------------------------------------------------------------------------
__global__ void __launch_bounds__(256)
sage_combine(const float* __restrict__ agg, const float* __restrict__ cnt,
             const float* __restrict__ X,
             const __bf16* __restrict__ Wl, const float* __restrict__ bl,
             const __bf16* __restrict__ Wr,
             float* __restrict__ out_f32, __bf16* __restrict__ out_bf16, int relu) {
  __shared__ __bf16 sA[16 * HH];
  __shared__ __bf16 sX[16 * HH];
  const int row0 = blockIdx.x * 16;
  const int tid = threadIdx.x;
  for (int i = tid; i < 16 * HH; i += 256) {
    const int r = i >> 7, c = i & 127;
    const int gr = row0 + r;
    const float scale = 1.0f / fmaxf(cnt[gr], 1.0f);
    sA[i] = (__bf16)(agg[(long long)gr * HH + c] * scale);
    sX[i] = (__bf16)(X[(long long)gr * HH + c]);
  }
  __syncthreads();

  const int wave = tid >> 5, lane = tid & 31;
  const int col = wave * 16 + (lane & 15);
  const int mh = (lane & 16) ? 8 : 0;

  v8f c;
  const float bias = bl[col];
#pragma unroll
  for (int r = 0; r < 8; ++r) c[r] = bias;

#pragma unroll
  for (int kt = 0; kt < 4; ++kt) {
    v16bf a0 = load_a_frag_lds(sA, HH, lane, kt * 32);
    v16bf b0 = load_b_frag_row(Wl + col * HH, lane, kt * 32);
    c = WMMA_BF16(a0, b0, c);
    v16bf a1 = load_a_frag_lds(sX, HH, lane, kt * 32);
    v16bf b1 = load_b_frag_row(Wr + col * HH, lane, kt * 32);
    c = WMMA_BF16(a1, b1, c);
  }

#pragma unroll
  for (int r = 0; r < 8; ++r) {
    float v = c[r];
    if (relu) v = fmaxf(v, 0.0f);
    const long long idx = (long long)(row0 + mh + r) * HH + col;
    if (out_bf16) out_bf16[idx] = (__bf16)v;
    else          out_f32[idx]  = v;
  }
}

// ---------------------------------------------------------------------------
// Fused edge decoder: async-gather [zu|zm] (bf16) -> 256 feat -> 512 (ReLU)
// -> 256 (ReLU) -> dot 256 -> scalar.  64 edges per block (4 M-tiles), 8
// waves.  B fragments are loaded once per (col-tile, k-step) and reused
// across all 4 M-tiles; all intermediates live in LDS only.
// LDS plan:  [0,32K)  sF  bf16[64][256]  -- aliased by sH2 bf16[64][256]
//            [32K,96K) sH1 bf16[64][512]
//            [96K, +1K) sRed f32[64][4]
// ---------------------------------------------------------------------------
__global__ void __launch_bounds__(256)
edge_decoder(const __bf16* __restrict__ zu, const __bf16* __restrict__ zm,
             const int* __restrict__ el_u, const int* __restrict__ el_m,
             const __bf16* __restrict__ W1, const float* __restrict__ b1,
             const __bf16* __restrict__ W2, const float* __restrict__ b2,
             const float* __restrict__ W3, const float* __restrict__ b3,
             float* __restrict__ out) {
  __shared__ __align__(16) char smem[(32 + 64) * 1024 + DEC_M * 4 * sizeof(float)];
  __bf16* sF   = (__bf16*)smem;                  // 64 x 256 bf16
  __bf16* sH1  = (__bf16*)(smem + 32 * 1024);    // 64 x 512 bf16
  __bf16* sH2  = (__bf16*)smem;                  // aliases sF (dead after stage 1)
  float*  sRed = (float*)(smem + 96 * 1024);

  const int e0 = blockIdx.x * DEC_M;
  const int tid = threadIdx.x;

  // ---- gather via async global->LDS copies (16B chunks, ASYNCcnt-tracked) --
  // feature row r = [ zu[el_u[e]] (128 bf16) | zm[el_m[e]] (128 bf16) ]
  for (int ch = tid; ch < DEC_M * 32; ch += 256) {   // 32 x 16B chunks per row
    const int r = ch >> 5, q = ch & 31;
    int e = e0 + r; if (e >= N_L) e = N_L - 1;       // clamp pad rows
    const __bf16* src = (q < 16) ? (zu + (long long)el_u[e] * HH + q * 8)
                                 : (zm + (long long)el_m[e] * HH + (q - 16) * 8);
    const unsigned ldsOff = (unsigned)(unsigned long long)(sF + r * 256 + q * 8);
    asm volatile("global_load_async_to_lds_b128 %0, %1, off"
                 :: "v"(ldsOff), "v"(src) : "memory");
  }
  asm volatile("s_wait_asynccnt 0" ::: "memory");
  __syncthreads();

  const int wave = tid >> 5, lane = tid & 31;
  const int nl = lane & 15;
  const int mh = (lane & 16) ? 8 : 0;

  // ---- stage 1: [64x256] @ [256x512]^T, each wave owns 4 column tiles -----
  for (int ctj = 0; ctj < 4; ++ctj) {
    const int col = (wave * 4 + ctj) * 16 + nl;
    v8f acc[4];
    const float bias = b1[col];
#pragma unroll
    for (int mt = 0; mt < 4; ++mt)
#pragma unroll
      for (int r = 0; r < 8; ++r) acc[mt][r] = bias;
#pragma unroll
    for (int kt = 0; kt < 8; ++kt) {
      const v16bf b = load_b_frag_row(W1 + col * 256, lane, kt * 32);
#pragma unroll
      for (int mt = 0; mt < 4; ++mt) {
        const v16bf a = load_a_frag_lds(sF + mt * 16 * 256, 256, lane, kt * 32);
        acc[mt] = WMMA_BF16(a, b, acc[mt]);
      }
    }
#pragma unroll
    for (int mt = 0; mt < 4; ++mt)
#pragma unroll
      for (int r = 0; r < 8; ++r)
        sH1[(mt * 16 + mh + r) * 512 + col] = (__bf16)fmaxf(acc[mt][r], 0.0f);
  }
  __syncthreads();

  // ---- stage 2: [64x512] @ [512x256]^T, each wave owns 2 column tiles -----
  for (int ctj = 0; ctj < 2; ++ctj) {
    const int col = (wave * 2 + ctj) * 16 + nl;
    v8f acc[4];
    const float bias = b2[col];
#pragma unroll
    for (int mt = 0; mt < 4; ++mt)
#pragma unroll
      for (int r = 0; r < 8; ++r) acc[mt][r] = bias;
#pragma unroll
    for (int kt = 0; kt < 16; ++kt) {
      const v16bf b = load_b_frag_row(W2 + col * 512, lane, kt * 32);
#pragma unroll
      for (int mt = 0; mt < 4; ++mt) {
        const v16bf a = load_a_frag_lds(sH1 + mt * 16 * 512, 512, lane, kt * 32);
        acc[mt] = WMMA_BF16(a, b, acc[mt]);
      }
    }
#pragma unroll
    for (int mt = 0; mt < 4; ++mt)
#pragma unroll
      for (int r = 0; r < 8; ++r)
        sH2[(mt * 16 + mh + r) * 256 + col] = (__bf16)fmaxf(acc[mt][r], 0.0f);
  }
  __syncthreads();

  // ---- stage 3: per-edge dot with W3 (f32 accumulate) ---------------------
  {
    const int i = tid >> 2;         // edge row 0..63
    const int q = tid & 3;          // quarter of K
    const int k0 = q * 64;
    float p = 0.0f;
    for (int k = 0; k < 64; ++k)
      p = fmaf((float)sH2[i * 256 + k0 + k], W3[k0 + k], p);
    sRed[i * 4 + q] = p;
  }
  __syncthreads();
  if (tid < DEC_M) {
    const float s = b3[0] + sRed[tid * 4 + 0] + sRed[tid * 4 + 1]
                          + sRed[tid * 4 + 2] + sRed[tid * 4 + 3];
    const int e = e0 + tid;
    if (e < N_L) out[e] = s;
  }
}

// ---------------------------------------------------------------------------
// Host-side orchestration
// ---------------------------------------------------------------------------
extern "C" void kernel_launch(void* const* d_in, const int* in_sizes, int n_in,
                              void* d_out, int out_size, void* d_ws, size_t ws_size,
                              hipStream_t stream) {
  (void)in_sizes; (void)n_in; (void)out_size; (void)ws_size;

  const float* x_user  = (const float*)d_in[0];
  const float* x_movie = (const float*)d_in[1];
  const int*   eidx    = (const int*)d_in[2];
  const int*   elidx   = (const int*)d_in[3];
  const float* lin_u_W = (const float*)d_in[4];
  const float* lin_u_b = (const float*)d_in[5];
  const float* lin_m_W = (const float*)d_in[6];
  const float* lin_m_b = (const float*)d_in[7];
  const float* bn_u_g = (const float*)d_in[8];  const float* bn_u_b = (const float*)d_in[9];
  const float* bn_u_m = (const float*)d_in[10]; const float* bn_u_v = (const float*)d_in[11];
  const float* bn_m_g = (const float*)d_in[12]; const float* bn_m_b = (const float*)d_in[13];
  const float* bn_m_m = (const float*)d_in[14]; const float* bn_m_v = (const float*)d_in[15];
  const float* c1_um_Wl = (const float*)d_in[16]; const float* c1_um_bl = (const float*)d_in[17];
  const float* c1_um_Wr = (const float*)d_in[18];
  const float* c1_mu_Wl = (const float*)d_in[19]; const float* c1_mu_bl = (const float*)d_in[20];
  const float* c1_mu_Wr = (const float*)d_in[21];
  const float* c2_um_Wl = (const float*)d_in[22]; const float* c2_um_bl = (const float*)d_in[23];
  const float* c2_um_Wr = (const float*)d_in[24];
  const float* c2_mu_Wl = (const float*)d_in[25]; const float* c2_mu_bl = (const float*)d_in[26];
  const float* c2_mu_Wr = (const float*)d_in[27];
  const float* dec_W1 = (const float*)d_in[28]; const float* dec_b1 = (const float*)d_in[29];
  const float* dec_W2 = (const float*)d_in[30]; const float* dec_b2 = (const float*)d_in[31];
  const float* dec_W3 = (const float*)d_in[32]; const float* dec_b3 = (const float*)d_in[33];

  const int* e_u  = eidx;           // edge_index[0]
  const int* e_m  = eidx + N_E;     // edge_index[1]
  const int* el_u = elidx;          // edge_label_index[0]
  const int* el_m = elidx + N_L;    // edge_label_index[1]

  // ---- workspace layout ----
  char* ws = (char*)d_ws;
  size_t off = 0;
  auto take = [&](size_t bytes) { char* p = ws + off; off += (bytes + 255) & ~(size_t)255; return p; };
  float* XU   = (float*)take((size_t)N_U * HH * 4);
  float* XM   = (float*)take((size_t)N_M * HH * 4);
  float* XU1  = (float*)take((size_t)N_U * HH * 4);
  float* XM1  = (float*)take((size_t)N_M * HH * 4);
  float* AGGU = (float*)take((size_t)N_U * HH * 4);   // one zero pass; CNTU pad leaves CNTM tail unzeroed
  float* AGGM = (float*)take((size_t)N_M * HH * 4);
  float* CNTU = (float*)take((size_t)N_U * 4);
  float* CNTM = (float*)take((size_t)N_M * 4);
  __bf16* ZUb = (__bf16*)take((size_t)N_U * HH * 2);  // conv2 outputs in bf16 (decoder input)
  __bf16* ZMb = (__bf16*)take((size_t)N_M * HH * 2);
  __bf16* W_c1um_l = (__bf16*)take((size_t)HH * HH * 2);
  __bf16* W_c1um_r = (__bf16*)take((size_t)HH * HH * 2);
  __bf16* W_c1mu_l = (__bf16*)take((size_t)HH * HH * 2);
  __bf16* W_c1mu_r = (__bf16*)take((size_t)HH * HH * 2);
  __bf16* W_c2um_l = (__bf16*)take((size_t)HH * HH * 2);
  __bf16* W_c2um_r = (__bf16*)take((size_t)HH * HH * 2);
  __bf16* W_c2mu_l = (__bf16*)take((size_t)HH * HH * 2);
  __bf16* W_c2mu_r = (__bf16*)take((size_t)HH * HH * 2);
  __bf16* W1b = (__bf16*)take((size_t)512 * 256 * 2);
  __bf16* W2b = (__bf16*)take((size_t)256 * 512 * 2);

  // ---- weight conversion to bf16 ----
  {
    const int nw = HH * HH;
    cvt_f32_bf16<<<(nw + 255) / 256, 256, 0, stream>>>(c1_um_Wl, W_c1um_l, nw);
    cvt_f32_bf16<<<(nw + 255) / 256, 256, 0, stream>>>(c1_um_Wr, W_c1um_r, nw);
    cvt_f32_bf16<<<(nw + 255) / 256, 256, 0, stream>>>(c1_mu_Wl, W_c1mu_l, nw);
    cvt_f32_bf16<<<(nw + 255) / 256, 256, 0, stream>>>(c1_mu_Wr, W_c1mu_r, nw);
    cvt_f32_bf16<<<(nw + 255) / 256, 256, 0, stream>>>(c2_um_Wl, W_c2um_l, nw);
    cvt_f32_bf16<<<(nw + 255) / 256, 256, 0, stream>>>(c2_um_Wr, W_c2um_r, nw);
    cvt_f32_bf16<<<(nw + 255) / 256, 256, 0, stream>>>(c2_mu_Wl, W_c2mu_l, nw);
    cvt_f32_bf16<<<(nw + 255) / 256, 256, 0, stream>>>(c2_mu_Wr, W_c2mu_r, nw);
    cvt_f32_bf16<<<(512 * 256 + 255) / 256, 256, 0, stream>>>(dec_W1, W1b, 512 * 256);
    cvt_f32_bf16<<<(256 * 512 + 255) / 256, 256, 0, stream>>>(dec_W2, W2b, 256 * 512);
  }

  // ---- input projections + BN + ReLU ----
  proj_bn_relu<<<N_U, 128, 0, stream>>>(x_user, lin_u_W, lin_u_b,
                                        bn_u_g, bn_u_b, bn_u_m, bn_u_v, XU, F_U);
  proj_bn_relu<<<N_M, 128, 0, stream>>>(x_movie, lin_m_W, lin_m_b,
                                        bn_m_g, bn_m_b, bn_m_m, bn_m_v, XM, F_M);

  const long long aggTotal = (long long)N_U * HH + (long long)N_M * HH + N_U + N_M;
  const long long scatThreads = (long long)N_E * 32;
  const int scatBlocks = (int)((scatThreads + 255) / 256);

  // ---- conv1 ----
  zero_f32<<<(int)((aggTotal + 255) / 256), 256, 0, stream>>>(AGGU, aggTotal);
  scatter_edges<<<scatBlocks, 256, 0, stream>>>(XU, e_u, e_m, AGGM, CNTM, N_E);
  scatter_edges<<<scatBlocks, 256, 0, stream>>>(XM, e_m, e_u, AGGU, CNTU, N_E);
  sage_combine<<<N_M / 16, 256, 0, stream>>>(AGGM, CNTM, XM, W_c1um_l, c1_um_bl, W_c1um_r,
                                             XM1, (__bf16*)nullptr, 1);
  sage_combine<<<N_U / 16, 256, 0, stream>>>(AGGU, CNTU, XU, W_c1mu_l, c1_mu_bl, W_c1mu_r,
                                             XU1, (__bf16*)nullptr, 1);

  // ---- conv2 (outputs bf16 zu/zm for the decoder) ----
  zero_f32<<<(int)((aggTotal + 255) / 256), 256, 0, stream>>>(AGGU, aggTotal);
  scatter_edges<<<scatBlocks, 256, 0, stream>>>(XU1, e_u, e_m, AGGM, CNTM, N_E);
  scatter_edges<<<scatBlocks, 256, 0, stream>>>(XM1, e_m, e_u, AGGU, CNTU, N_E);
  sage_combine<<<N_M / 16, 256, 0, stream>>>(AGGM, CNTM, XM1, W_c2um_l, c2_um_bl, W_c2um_r,
                                             (float*)nullptr, ZMb, 0);
  sage_combine<<<N_U / 16, 256, 0, stream>>>(AGGU, CNTU, XU1, W_c2mu_l, c2_mu_bl, W_c2mu_r,
                                             (float*)nullptr, ZUb, 0);

  // ---- fused edge decoder ----
  edge_decoder<<<(N_L + DEC_M - 1) / DEC_M, 256, 0, stream>>>(
      ZUb, ZMb, el_u, el_m, W1b, dec_b1, W2b, dec_b2, dec_W3, dec_b3, (float*)d_out);
}
